// GlobalLinearAttention_Sparse_77799037599834
// MI455X (gfx1250) — matrix operand order repaired and checked by reading, not compiled, not run on hardware
//
#include <hip/hip_runtime.h>

// ---------------------------------------------------------------------------
// Problem constants (from reference)
// ---------------------------------------------------------------------------
#define GRP   128
#define NPG   256
#define DIM   512
#define HEADS 8
#define DH    64
#define INNER 512          // HEADS*DH
#define NROW  (GRP * NPG)  // 32768
#define FFD   2048         // 4*DIM
#define EPSV  1e-5f

typedef __attribute__((ext_vector_type(16))) __bf16 v16bf;
typedef __attribute__((ext_vector_type(8)))  float  v8f;
typedef __attribute__((ext_vector_type(8)))  __bf16 bf16x8;
typedef __attribute__((ext_vector_type(2)))  __bf16 bf16x2;

// ---------------------------------------------------------------------------
// CDNA5 async global->LDS copy (VGLOBAL GLOBAL_LOAD_ASYNC_TO_LDS_B128, GV mode)
// vdst = LDS byte address VGPR, vaddr = 64-bit global address. Tracked by
// ASYNCcnt; completion enforced with s_wait_asynccnt.
// ---------------------------------------------------------------------------
__device__ inline void async_copy_b128(unsigned lds_byte_off, const void* gaddr) {
    asm volatile("global_load_async_to_lds_b128 %0, %1, off"
                 :: "v"(lds_byte_off), "v"((unsigned long long)(uintptr_t)gaddr)
                 : "memory");
}
__device__ inline void wait_async_le6() { asm volatile("s_wait_asynccnt 0x6" ::: "memory"); }
__device__ inline void wait_async_0()   { asm volatile("s_wait_asynccnt 0x0" ::: "memory"); }

// ---------------------------------------------------------------------------
// WMMA fragment loaders (CDNA5 ISA 7.12.2 bf16 layouts, wave32)
// A 16x32 (MxK): lane m=lane&15, khalf=lane>>4; vgpr p holds K = 16*(p/4)+8*khalf+2*(p%4) .. +1
// B 32x16 (KxN) staged K-contiguous per n ("Bt[n][k]"): lane n=lane&15, khalf=lane>>4;
//   vgpr p holds K = 16*(p/4)+8*khalf+2*(p%4) .. +1
// C 16x16 f32: vgpr r, lane l -> m = r + 8*(l>>4), n = l&15
// Both gathers are 2x16B contiguous per lane -> ds_load_b128 pairs.
// ---------------------------------------------------------------------------
__device__ inline v16bf load_a_frag(const __bf16* __restrict__ A, int lda, int lane) {
    int m = lane & 15;
    int khalf = lane >> 4;
    v16bf r;
#pragma unroll
    for (int p = 0; p < 8; ++p) {
        int k = ((p >> 2) << 4) + (khalf << 3) + ((p & 3) << 1);
        r[2 * p]     = A[m * lda + k];
        r[2 * p + 1] = A[m * lda + k + 1];
    }
    return r;
}

__device__ inline v16bf load_b_frag(const __bf16* __restrict__ Bt, int ldb, int lane) {
    int n = lane & 15;
    int khalf = lane >> 4;
    v16bf r;
#pragma unroll
    for (int p = 0; p < 8; ++p) {
        int k = ((p >> 2) << 4) + (khalf << 3) + ((p & 3) << 1);
        r[2 * p]     = Bt[n * ldb + k];
        r[2 * p + 1] = Bt[n * ldb + k + 1];
    }
    return r;
}

// ---------------------------------------------------------------------------
// fp32 weight [K][N] -> bf16 transposed [N][K]
// ---------------------------------------------------------------------------
__global__ void cvt_t_bf16_kernel(const float* __restrict__ in, __bf16* __restrict__ out,
                                  int K, int N) {
    int i = blockIdx.x * blockDim.x + threadIdx.x;
    if (i < K * N) {
        int k = i / N, n = i % N;
        out[(size_t)n * K + k] = (__bf16)in[i];
    }
}

// ---------------------------------------------------------------------------
// Graph layernorm: per-group mean / rstd over NPG*DIM elements
// ---------------------------------------------------------------------------
__global__ __launch_bounds__(256) void ln_stats_kernel(const float* __restrict__ x,
                                                       float* __restrict__ stats) {
    __shared__ float ssum[256], ssq[256];
    int g = blockIdx.x, tid = threadIdx.x;
    const float* p = x + (size_t)g * (NPG * DIM);
    float s = 0.f, q = 0.f;
    for (int i = tid; i < NPG * DIM; i += 256) {
        float v = p[i];
        s += v; q += v * v;
    }
    ssum[tid] = s; ssq[tid] = q;
    __syncthreads();
    for (int off = 128; off > 0; off >>= 1) {
        if (tid < off) { ssum[tid] += ssum[tid + off]; ssq[tid] += ssq[tid + off]; }
        __syncthreads();
    }
    if (tid == 0) {
        const float inv = 1.0f / (float)(NPG * DIM);
        float mean = ssum[0] * inv;
        float var  = ssq[0] * inv - mean * mean;
        stats[2 * g]     = mean;
        stats[2 * g + 1] = rsqrtf(var + EPSV);
    }
}

__global__ void ln_apply_kernel(const float* __restrict__ x, const float* __restrict__ stats,
                                const float* __restrict__ w, const float* __restrict__ b,
                                float* __restrict__ outF, __bf16* __restrict__ outB) {
    size_t i = (size_t)blockIdx.x * blockDim.x + threadIdx.x;
    int g   = (int)(i / (size_t)(NPG * DIM));
    int col = (int)(i % DIM);
    float m = stats[2 * g], r = stats[2 * g + 1];
    float v = (x[i] - m) * r * w[col] + b[col];
    if (outF) outF[i] = v;
    if (outB) outB[i] = (__bf16)v;
}

__global__ void add_kernel(const float* __restrict__ a, const float* __restrict__ b,
                           float* __restrict__ out) {
    size_t i = (size_t)blockIdx.x * blockDim.x + threadIdx.x;
    out[i] = a[i] + b[i];
}

// ---------------------------------------------------------------------------
// bf16 WMMA GEMM: C[M,Nc] = A[M,K] @ Bt[Nc,K]^T  (+bias) (gelu) (+residual)
// Block tile 256x128, BK=32, 256 threads = 8 waves (4x2), wave tile 64x64
// -> 16 v_wmma_f32_16x16x32_bf16 per K-chunk from 16 ds_load_b128.
// Double-buffered LDS filled with GLOBAL_LOAD_ASYNC_TO_LDS_B128 (ASYNCcnt),
// 6 async b128 per thread per stage, pipelined with s_wait_asynccnt 6.
// Requires M%256==0, Nc%128==0, K%32==0 (true for all calls here).
// ---------------------------------------------------------------------------
__global__ __launch_bounds__(256) void gemm_bf16_kernel(
    const __bf16* __restrict__ A, const __bf16* __restrict__ Bt,
    float* __restrict__ Cf, __bf16* __restrict__ Cb,
    const float* __restrict__ bias, const float* __restrict__ resid,
    int M, int K, int Nc, int act) {
    constexpr int BM = 256, BN = 128, BK = 32, LSTR = BK + 8;  // 80B rows: 16B aligned, conflict-free
    __shared__ __bf16 As[2 * BM * LSTR];   // 40960 B
    __shared__ __bf16 Bs[2 * BN * LSTR];   // 20480 B

    const int tid  = threadIdx.x;
    const int lane = tid & 31;
    const int wave = tid >> 5;
    const int wm = wave >> 1, wn = wave & 1;          // 4 x 2 wave grid, wave tile 64x64
    const int blockRow = blockIdx.y * BM;
    const int blockCol = blockIdx.x * BN;

    // A staging: one full 32-element row (64B = 4 x b128) per thread
    const int arow = tid;                     // 0..255
    // B staging: half-row (16 elements = 2 x b128) per thread
    const int brow = tid >> 1;                // 0..127
    const int bkp  = (tid & 1) << 4;          // 0 or 16
    const __bf16* gA = A  + (size_t)(blockRow + arow) * K;
    const __bf16* gB = Bt + (size_t)(blockCol + brow) * K + bkp;

    v8f acc[4][4] = {};

    auto stage = [&](int k0, int buf) {
        unsigned la = (unsigned)(uintptr_t)&As[buf * (BM * LSTR) + arow * LSTR];
        unsigned lb = (unsigned)(uintptr_t)&Bs[buf * (BN * LSTR) + brow * LSTR + bkp];
        const __bf16* pa = gA + k0;
#pragma unroll
        for (int v = 0; v < 4; ++v) async_copy_b128(la + 16 * v, pa + 8 * v);
        async_copy_b128(lb,      gB + k0);
        async_copy_b128(lb + 16, gB + k0 + 8);
    };

    const int nchunk = K / BK;
    stage(0, 0);
    for (int c = 0; c < nchunk; ++c) {
        int buf = c & 1;
        if (c + 1 < nchunk) {
            stage((c + 1) * BK, buf ^ 1);
            wait_async_le6();                 // current stage's 6 copies retired (in-order)
        } else {
            wait_async_0();
        }
        __syncthreads();

        const __bf16* Asb = &As[buf * (BM * LSTR)];
        const __bf16* Bsb = &Bs[buf * (BN * LSTR)];
        v16bf aF[4], bF[4];
#pragma unroll
        for (int mt = 0; mt < 4; ++mt)
            aF[mt] = load_a_frag(Asb + (wm * 64 + mt * 16) * LSTR, LSTR, lane);
#pragma unroll
        for (int nt = 0; nt < 4; ++nt)
            bF[nt] = load_b_frag(Bsb + (wn * 64 + nt * 16) * LSTR, LSTR, lane);
#pragma unroll
        for (int mt = 0; mt < 4; ++mt)
#pragma unroll
            for (int nt = 0; nt < 4; ++nt)
                acc[mt][nt] = __builtin_amdgcn_wmma_f32_16x16x32_bf16(
                    false, aF[mt], false, bF[nt], (short)0, acc[mt][nt], false, false);
        __syncthreads();
    }

    // ---- epilogue: bias -> gelu(exact) -> residual -> store f32/bf16 ----
    const int lm  = (lane >> 4) << 3;
    const int ln_ = lane & 15;
#pragma unroll
    for (int mt = 0; mt < 4; ++mt)
#pragma unroll
        for (int nt = 0; nt < 4; ++nt)
#pragma unroll
            for (int r = 0; r < 8; ++r) {
                int m = blockRow + wm * 64 + mt * 16 + lm + r;
                int n = blockCol + wn * 64 + nt * 16 + ln_;
                float v = acc[mt][nt][r];
                if (bias)  v += bias[n];
                if (act)   v = 0.5f * v * (1.0f + erff(v * 0.70710678118f));
                size_t idx = (size_t)m * Nc + n;
                if (resid) v += resid[idx];
                if (Cf) Cf[idx] = v;
                if (Cb) Cb[idx] = (__bf16)v;
            }
}

// ---------------------------------------------------------------------------
// Attention: one block per (group, head). Q/K in LDS row-major [i][d] (async
// staged), V transposed [d][j], probability tile [i][j] in LDS bf16.
// S = QK^T/8 via WMMA -> per-row softmax -> O = P V via WMMA.
// ---------------------------------------------------------------------------
#define QSTR 72   // 64 + pad: 144B rows (16B aligned), lane stride 36 dwords (conflict-free)
#define VSTR 270  // 256 + pad (odd dword stride)

__global__ __launch_bounds__(256) void attn_kernel(const __bf16* __restrict__ qbuf,
                                                   const __bf16* __restrict__ kvbuf,
                                                   __bf16* __restrict__ obuf) {
    extern __shared__ __align__(16) char smem_raw[];
    __bf16* Qs = (__bf16*)smem_raw;            // 256*QSTR
    __bf16* Ks = Qs + 256 * QSTR;              // 256*QSTR
    __bf16* Vt = Ks + 256 * QSTR;              // 64*VSTR
    __bf16* Ps = Vt + 64 * VSTR;               // 256*VSTR

    const int g = blockIdx.x / HEADS;
    const int h = blockIdx.x % HEADS;
    const size_t base = (size_t)g * NPG;
    const int tid  = threadIdx.x;
    const int lane = tid & 31;
    const int wave = tid >> 5;
    const int m0 = wave * 32;                  // 8 waves x 32 rows

    // ---- stage Q, K (async b128) and V^T (register transpose) ----
    {
        const int i  = tid >> 3;               // 32 rows per pass
        const int dp = (tid & 7) << 3;         // 8 bf16 = 16B per thread
#pragma unroll
        for (int pass = 0; pass < 8; ++pass) {
            int ii = i + pass * 32;
            const __bf16* gq = qbuf  + (base + ii) * INNER       + h * DH + dp;
            const __bf16* gk = kvbuf + (base + ii) * (2 * INNER) + h * DH + dp;
            async_copy_b128((unsigned)(uintptr_t)&Qs[ii * QSTR + dp], gq);
            async_copy_b128((unsigned)(uintptr_t)&Ks[ii * QSTR + dp], gk);
            bf16x8 vv = *(const bf16x8*)(kvbuf + (base + ii) * (2 * INNER) + INNER + h * DH + dp);
#pragma unroll
            for (int e = 0; e < 8; ++e) Vt[(dp + e) * VSTR + ii] = vv[e];
        }
    }
    wait_async_0();
    __syncthreads();

    // ---- S = Q K^T * DH^-0.5 ----
    {
        v16bf aF[2][2];
#pragma unroll
        for (int mt = 0; mt < 2; ++mt)
#pragma unroll
            for (int kc = 0; kc < 2; ++kc)
                aF[mt][kc] = load_a_frag(Qs + (m0 + mt * 16) * QSTR + kc * 32, QSTR, lane);
        for (int jt = 0; jt < 16; ++jt) {
            v16bf bF0 = load_b_frag(Ks + (jt * 16) * QSTR, QSTR, lane);
            v16bf bF1 = load_b_frag(Ks + (jt * 16) * QSTR + 32, QSTR, lane);
#pragma unroll
            for (int mt = 0; mt < 2; ++mt) {
                v8f acc = {};
                acc = __builtin_amdgcn_wmma_f32_16x16x32_bf16(false, aF[mt][0], false, bF0,
                                                              (short)0, acc, false, false);
                acc = __builtin_amdgcn_wmma_f32_16x16x32_bf16(false, aF[mt][1], false, bF1,
                                                              (short)0, acc, false, false);
                int lm = (lane >> 4) << 3, ln_ = lane & 15;
#pragma unroll
                for (int r = 0; r < 8; ++r) {
                    int i = m0 + mt * 16 + lm + r;
                    int j = jt * 16 + ln_;
                    Ps[i * VSTR + j] = (__bf16)(acc[r] * 0.125f);
                }
            }
        }
    }
    __syncthreads();

    // ---- row softmax: one row per thread, bf16x2 (ds b32) accesses ----
    {
        const int i = tid;
        float mx = -3.0e38f;
        for (int j = 0; j < NPG; j += 2) {
            bf16x2 p = *(const bf16x2*)&Ps[i * VSTR + j];
            mx = fmaxf(mx, fmaxf((float)p.x, (float)p.y));
        }
        float sum = 0.f;
        for (int j = 0; j < NPG; j += 2) {
            bf16x2 p = *(const bf16x2*)&Ps[i * VSTR + j];
            sum += __expf((float)p.x - mx) + __expf((float)p.y - mx);
        }
        float inv = 1.0f / sum;
        for (int j = 0; j < NPG; j += 2) {
            bf16x2 p = *(const bf16x2*)&Ps[i * VSTR + j];
            bf16x2 o;
            o.x = (__bf16)(__expf((float)p.x - mx) * inv);
            o.y = (__bf16)(__expf((float)p.y - mx) * inv);
            *(bf16x2*)&Ps[i * VSTR + j] = o;
        }
    }
    __syncthreads();

    // ---- O = P V ----
#pragma unroll
    for (int mt = 0; mt < 2; ++mt) {
        v8f acc[4] = {};
        for (int kc = 0; kc < 8; ++kc) {
            v16bf aF = load_a_frag(Ps + (m0 + mt * 16) * VSTR + kc * 32, VSTR, lane);
#pragma unroll
            for (int nt = 0; nt < 4; ++nt) {
                v16bf bF = load_b_frag(Vt + (nt * 16) * VSTR + kc * 32, VSTR, lane);
                acc[nt] = __builtin_amdgcn_wmma_f32_16x16x32_bf16(false, aF, false, bF,
                                                                  (short)0, acc[nt], false, false);
            }
        }
        int lm = (lane >> 4) << 3, ln_ = lane & 15;
#pragma unroll
        for (int nt = 0; nt < 4; ++nt)
#pragma unroll
            for (int r = 0; r < 8; ++r) {
                int i = m0 + mt * 16 + lm + r;
                int d = nt * 16 + ln_;
                obuf[(base + i) * INNER + h * DH + d] = (__bf16)acc[nt][r];
            }
    }
}

// ---------------------------------------------------------------------------
// Host launcher
// ---------------------------------------------------------------------------
static inline void gemm_launch(hipStream_t s, const __bf16* A, const __bf16* Bt,
                               float* Cf, __bf16* Cb, const float* bias,
                               const float* resid, int M, int K, int Nc, int act) {
    dim3 grid(Nc / 128, M / 256);
    gemm_bf16_kernel<<<grid, 256, 0, s>>>(A, Bt, Cf, Cb, bias, resid, M, K, Nc, act);
}

extern "C" void kernel_launch(void* const* d_in, const int* in_sizes, int n_in,
                              void* d_out, int out_size, void* d_ws, size_t ws_size,
                              hipStream_t stream) {
    const float* x        = (const float*)d_in[0];
    const float* queries  = (const float*)d_in[1];
    // d_in[2] batch (int64) unused — rows are contiguous per group
    const float* ln_seq_w = (const float*)d_in[3];
    const float* ln_seq_b = (const float*)d_in[4];
    const float* ln_q_w   = (const float*)d_in[5];
    const float* ln_q_b   = (const float*)d_in[6];
    const float* a1_Wq    = (const float*)d_in[7];
    const float* a1_Wkv   = (const float*)d_in[8];
    const float* a1_Wo    = (const float*)d_in[9];
    const float* a1_bo    = (const float*)d_in[10];
    const float* a2_Wq    = (const float*)d_in[11];
    const float* a2_Wkv   = (const float*)d_in[12];
    const float* a2_Wo    = (const float*)d_in[13];
    const float* a2_bo    = (const float*)d_in[14];
    const float* ff_ln_w  = (const float*)d_in[15];
    const float* ff_ln_b  = (const float*)d_in[16];
    const float* ff_W1    = (const float*)d_in[17];
    const float* ff_b1    = (const float*)d_in[18];
    const float* ff_W2    = (const float*)d_in[19];
    const float* ff_b2    = (const float*)d_in[20];

    float* out_x = (float*)d_out;                         // x_out: N*DIM
    float* out_q = (float*)d_out + (size_t)NROW * DIM;    // q2:    N*DIM

    // ---- bump allocator on workspace ----
    char* wsp = (char*)d_ws;
    auto alloc = [&](size_t bytes) -> void* {
        void* p = (void*)wsp;
        wsp += (bytes + 255) & ~(size_t)255;
        return p;
    };
    auto ab = [&](size_t n) { return (__bf16*)alloc(n * sizeof(__bf16)); };
    auto af = [&](size_t n) { return (float*)alloc(n * sizeof(float)); };

    const size_t ND = (size_t)NROW * DIM;
    // weights stored bf16 TRANSPOSED: Wt[N][K]
    __bf16* wq1  = ab((size_t)DIM * INNER);
    __bf16* wkv1 = ab((size_t)DIM * 2 * INNER);
    __bf16* wo1  = ab((size_t)INNER * DIM);
    __bf16* wq2  = ab((size_t)DIM * INNER);
    __bf16* wkv2 = ab((size_t)DIM * 2 * INNER);
    __bf16* wo2  = ab((size_t)INNER * DIM);
    __bf16* w1   = ab((size_t)DIM * FFD);
    __bf16* w2   = ab((size_t)FFD * DIM);

    __bf16* xn_b   = ab(ND);
    __bf16* qn_b   = ab(ND);
    __bf16* q1     = ab(ND);
    __bf16* kv1    = ab(2 * ND);
    __bf16* o1     = ab(ND);
    float*  ind_f  = af(ND);
    __bf16* ind_b  = ab(ND);
    __bf16* q2b    = ab(ND);
    __bf16* kv2    = ab(2 * ND);
    __bf16* o2     = ab(ND);
    float*  x2_f   = af(ND);
    float*  xn2_f  = af(ND);
    __bf16* xn2_b  = ab(ND);
    __bf16* h_b    = ab((size_t)NROW * FFD);
    float*  st0    = af(2 * GRP);
    float*  st1    = af(2 * GRP);
    float*  st2    = af(2 * GRP);

    // ---- convert weights to bf16, transposed to [N][K] ----
    auto cvt = [&](const float* src, __bf16* dst, int K, int N) {
        int n = K * N;
        cvt_t_bf16_kernel<<<(n + 255) / 256, 256, 0, stream>>>(src, dst, K, N);
    };
    cvt(a1_Wq, wq1, DIM, INNER);
    cvt(a1_Wkv, wkv1, DIM, 2 * INNER);
    cvt(a1_Wo, wo1, INNER, DIM);
    cvt(a2_Wq, wq2, DIM, INNER);
    cvt(a2_Wkv, wkv2, DIM, 2 * INNER);
    cvt(a2_Wo, wo2, INNER, DIM);
    cvt(ff_W1, w1, DIM, FFD);
    cvt(ff_W2, w2, FFD, DIM);

    const int LN_BLOCKS = (int)(ND / 256);

    // ---- graph layernorms of x and queries (bf16 copies only) ----
    ln_stats_kernel<<<GRP, 256, 0, stream>>>(x, st0);
    ln_apply_kernel<<<LN_BLOCKS, 256, 0, stream>>>(x, st0, ln_seq_w, ln_seq_b, nullptr, xn_b);
    ln_stats_kernel<<<GRP, 256, 0, stream>>>(queries, st1);
    ln_apply_kernel<<<LN_BLOCKS, 256, 0, stream>>>(queries, st1, ln_q_w, ln_q_b, nullptr, qn_b);

    const size_t attn_smem =
        (2 * (size_t)256 * QSTR + (size_t)64 * VSTR + (size_t)256 * VSTR) * sizeof(__bf16);
    hipFuncSetAttribute((const void*)attn_kernel,
                        hipFuncAttributeMaxDynamicSharedMemorySize, (int)attn_smem);

    // ---- attention 1: q from qn, kv from xn ----
    gemm_launch(stream, qn_b, wq1, nullptr, q1, nullptr, nullptr, NROW, DIM, INNER, 0);
    gemm_launch(stream, xn_b, wkv1, nullptr, kv1, nullptr, nullptr, NROW, DIM, 2 * INNER, 0);
    attn_kernel<<<GRP * HEADS, 256, attn_smem, stream>>>(q1, kv1, o1);
    gemm_launch(stream, o1, wo1, ind_f, ind_b, a1_bo, nullptr, NROW, INNER, DIM, 0);

    // q2 = induced + queries
    add_kernel<<<LN_BLOCKS, 256, 0, stream>>>(ind_f, queries, out_q);

    // ---- attention 2: q from xn, kv from induced ----
    gemm_launch(stream, xn_b, wq2, nullptr, q2b, nullptr, nullptr, NROW, DIM, INNER, 0);
    gemm_launch(stream, ind_b, wkv2, nullptr, kv2, nullptr, nullptr, NROW, DIM, 2 * INNER, 0);
    attn_kernel<<<GRP * HEADS, 256, attn_smem, stream>>>(q2b, kv2, o2);
    // x2 = o2 @ Wo2 + bo2 + x   (residual fused)
    gemm_launch(stream, o2, wo2, x2_f, nullptr, a2_bo, x, NROW, INNER, DIM, 0);

    // ---- FFN: x_norm = graphLN(x2); h = gelu(x_norm@W1+b1); out = h@W2+b2+x_norm ----
    ln_stats_kernel<<<GRP, 256, 0, stream>>>(x2_f, st2);
    ln_apply_kernel<<<LN_BLOCKS, 256, 0, stream>>>(x2_f, st2, ff_ln_w, ff_ln_b, xn2_f, xn2_b);
    gemm_launch(stream, xn2_b, w1, nullptr, h_b, ff_b1, nullptr, NROW, DIM, FFD, 1 /*gelu*/);
    gemm_launch(stream, h_b, w2, out_x, nullptr, ff_b2, xn2_f, NROW, FFD, DIM, 0);

    (void)in_sizes; (void)n_in; (void)out_size; (void)ws_size;
}
